// KMeans_23210003268101
// MI455X (gfx1250) — compile-verified, matrix operand order checked by reading, not run on hardware
//
#include <hip/hip_runtime.h>
#include <hip/hip_bf16.h>
#include <stdint.h>

// K-means (Lloyd) for MI455X / gfx1250, wave32.
// features [65536,1024] f32, logits [65536,256] f32 -> labels int64 [65536].
// Strategy: one-time bf16 copy of features (128MB, fits 192MB L2); assignment
// GEMM via v_wmma_f32_16x16x32_bf16: centroid K-chunk staged in LDS with
// double-buffered global_load_async_to_lds_b128 (ASYNCcnt), 4x4 fragment
// tiling per wave (B frag reused by 4 WMMAs); scatter-mean via f32 atomics.

#define WAVE    32
#define DIMS    1024
#define NCLASS  256
#define NROWS   65536
#define NITERS  25
#define NKCH    (DIMS / 32)      // 32 K-chunks of 32
#define BROW    40               // padded LDS row stride in bf16 elems (80 B)
#define RPB     128              // rows per block (8 waves: 2 row-grp x 4 class-grp)

typedef __attribute__((ext_vector_type(16))) __bf16 v16bf;
typedef __attribute__((ext_vector_type(8)))  float  v8f;

union FragBF {
  v16bf v;
  uint4 q[2];
  unsigned short h[16];
};

__device__ __forceinline__ unsigned short f32_to_bf16(float f) {
  unsigned int u = __float_as_uint(f);
  u += 0x7fffu + ((u >> 16) & 1u);   // round-to-nearest-even
  return (unsigned short)(u >> 16);
}
__device__ __forceinline__ float bf16_to_f32(unsigned short h) {
  return __uint_as_float(((unsigned int)h) << 16);
}

// ---- CDNA5 async copy: global memory -> LDS, tracked by ASYNCcnt -----------
__device__ __forceinline__ void async_copy_b128(unsigned lds_off, const void* gaddr) {
  asm volatile("global_load_async_to_lds_b128 %0, %1, off"
               :: "v"(lds_off), "v"(gaddr) : "memory");
}
template <int N>
__device__ __forceinline__ void wait_asynccnt() {
  asm volatile("s_wait_asynccnt %0" :: "n"(N) : "memory");
}

// ---------------- f32 -> bf16 bulk convert (8 elems / thread) ----------------
__global__ void convert_bf16(const float* __restrict__ x, unsigned short* __restrict__ xb) {
  size_t gid = (size_t)blockIdx.x * blockDim.x + threadIdx.x;
  size_t base = gid * 8;
  float4 a = *(const float4*)(x + base);
  float4 b = *(const float4*)(x + base + 4);
  uint4 o;
  o.x = (unsigned)f32_to_bf16(a.x) | ((unsigned)f32_to_bf16(a.y) << 16);
  o.y = (unsigned)f32_to_bf16(a.z) | ((unsigned)f32_to_bf16(a.w) << 16);
  o.z = (unsigned)f32_to_bf16(b.x) | ((unsigned)f32_to_bf16(b.y) << 16);
  o.w = (unsigned)f32_to_bf16(b.z) | ((unsigned)f32_to_bf16(b.w) << 16);
  *(uint4*)(xb + base) = o;
}

// ---------------- init labels: wave-per-row argmax over 256 logits ----------
__global__ void argmax_rows(const float* __restrict__ logits, unsigned int* __restrict__ labels) {
  int lane = threadIdx.x & (WAVE - 1);
  int row  = blockIdx.x * (blockDim.x / WAVE) + (threadIdx.x / WAVE);
  const float* p = logits + (size_t)row * NCLASS;
  float best = -__builtin_inff();
  int   bi   = 0;
#pragma unroll
  for (int j = 0; j < NCLASS / WAVE; ++j) {
    int c = lane + j * WAVE;     // increasing index -> strict > keeps first max
    float v = p[c];
    if (v > best) { best = v; bi = c; }
  }
#pragma unroll
  for (int m = 16; m >= 1; m >>= 1) {
    float ov = __shfl_xor(best, m, WAVE);
    int   oi = __shfl_xor(bi,   m, WAVE);
    if (ov > best || (ov == best && oi < bi)) { best = ov; bi = oi; }
  }
  if (lane == 0) labels[row] = (unsigned int)bi;
}

// ---------------- zero a float region ---------------------------------------
__global__ void zero_f32(float* __restrict__ p, int n) {
  int i = blockIdx.x * blockDim.x + threadIdx.x;
  if (i < n) p[i] = 0.0f;
}

// ---------------- scatter-add features into per-class sums ------------------
__global__ void accumulate_bf16(const unsigned short* __restrict__ xb,
                                const unsigned int* __restrict__ labels,
                                float* __restrict__ sums, float* __restrict__ counts) {
  int row = blockIdx.x;
  unsigned int lab = labels[row];
  int t = threadIdx.x;                       // 256 threads x 4 dims
  const ushort4 h = ((const ushort4*)(xb + (size_t)row * DIMS))[t];
  float* s = sums + (size_t)lab * DIMS + 4 * t;
  atomicAdd(s + 0, bf16_to_f32(h.x));
  atomicAdd(s + 1, bf16_to_f32(h.y));
  atomicAdd(s + 2, bf16_to_f32(h.z));
  atomicAdd(s + 3, bf16_to_f32(h.w));
  if (t == 0) atomicAdd(&counts[lab], 1.0f);
}

__global__ void accumulate_f32(const float* __restrict__ x,
                               const unsigned int* __restrict__ labels,
                               float* __restrict__ sums, float* __restrict__ counts) {
  int row = blockIdx.x;
  unsigned int lab = labels[row];
  int t = threadIdx.x;
  const float4 v = ((const float4*)(x + (size_t)row * DIMS))[t];
  float* s = sums + (size_t)lab * DIMS + 4 * t;
  atomicAdd(s + 0, v.x);
  atomicAdd(s + 1, v.y);
  atomicAdd(s + 2, v.z);
  atomicAdd(s + 3, v.w);
  if (t == 0) atomicAdd(&counts[lab], 1.0f);
}

// ---------------- centroids = sums/counts (keep prev if empty), norms -------
__global__ void finalize_centroids(const float* __restrict__ sums,
                                   const float* __restrict__ counts,
                                   float* __restrict__ centroids,        // prev in, new out
                                   unsigned short* __restrict__ cbuf,    // bf16 copy
                                   float* __restrict__ norms) {
  int c = blockIdx.x;
  int t = threadIdx.x;                       // 256 threads x 4 dims
  float cnt = counts[c];
  float inv = cnt > 0.0f ? 1.0f / cnt : 0.0f;
  float ns = 0.0f;
#pragma unroll
  for (int i = 0; i < 4; ++i) {
    size_t idx = (size_t)c * DIMS + 4 * t + i;
    float mu = (cnt > 0.0f) ? sums[idx] * inv : centroids[idx];
    centroids[idx] = mu;
    unsigned short hb = f32_to_bf16(mu);
    cbuf[idx] = hb;
    float mub = bf16_to_f32(hb);             // norm consistent with bf16 GEMM operand
    ns += mub * mub;
  }
  __shared__ float red[256];
  red[t] = ns;
  __syncthreads();
  for (int s = 128; s > 0; s >>= 1) {
    if (t < s) red[t] += red[t + s];
    __syncthreads();
  }
  if (t == 0) norms[c] = red[0];
}

// ---------------- assignment: WMMA bf16 GEMM + fused argmin -----------------
// Block = 8 waves = 128 rows x 256 classes. Wave tile: 64 rows x 64 classes
// (4 A-frags x 4 B-frags, 16 v8f accumulators). Centroid K-chunk (256x32 bf16)
// staged in LDS via double-buffered async copies; each B frag feeds 4 WMMAs so
// the ds-load dependency wait is amortized 4x. Per-row argmin finished with a
// cross-wave LDS reduction over the 4 class groups.
template <bool BF16SRC>
__global__ __launch_bounds__(256) void assign_kernel(
    const void* __restrict__ xsrc,
    const unsigned short* __restrict__ cb,   // [NCLASS, DIMS] bf16
    const float* __restrict__ norms,         // [NCLASS]
    unsigned int* __restrict__ labels) {
  __shared__ unsigned short btile[2][NCLASS * BROW];   // 2 x 20KB, padded rows
  __shared__ float sv[4][RPB];                         // per-classgroup best val
  __shared__ int   si[4][RPB];                         // per-classgroup best idx

  const int t    = threadIdx.x;
  const int lane = t & (WAVE - 1);
  const int l15  = lane & 15;
  const int half = lane >> 4;                // K base: 0 or 8 (ISA 16-bit A/B layout)
  const int wid  = t / WAVE;
  const int rg   = wid >> 2;                 // row group (0..1): 64 rows each
  const int cg   = wid & 3;                  // class group (0..3): 64 classes each
  const int brow0 = blockIdx.x * RPB;
  const int wrow0 = brow0 + rg * 64;         // wave's first row

  const unsigned short* xb = (const unsigned short*)xsrc;
  const float*          xf = (const float*)xsrc;

  // Per-lane A source for frag aq: row = wrow0 + aq*16 + l15,
  // Ks = half*8 + {0..7} and half*8 + 16 + {0..7}
  const size_t arow_off = (size_t)(wrow0 + l15) * DIMS + half * 8;

  // Async stage of one B k-chunk: thread t copies class row t (64B = 4 x b128).
  const unsigned short* bsrc_row = cb + (size_t)t * DIMS;
  const unsigned lds_row[2] = {
    (unsigned)(uintptr_t)&btile[0][t * BROW],
    (unsigned)(uintptr_t)&btile[1][t * BROW] };
  auto issue_btile = [&](int buf, int kc) {
#pragma unroll
    for (int j = 0; j < 4; ++j)
      async_copy_b128(lds_row[buf] + j * 16, bsrc_row + kc * 32 + j * 8);
  };

  v8f acc[16];                               // acc[aq*4 + nt]
#pragma unroll
  for (int i = 0; i < 16; ++i) acc[i] = (v8f){0.f, 0.f, 0.f, 0.f, 0.f, 0.f, 0.f, 0.f};

  issue_btile(0, 0);

  for (int kc = 0; kc < NKCH; ++kc) {
    const int k0  = kc * 32;
    const int cur = kc & 1;

    // 4 A fragments for this chunk (independent of LDS traffic; issue early).
    FragBF a[4];
#pragma unroll
    for (int aq = 0; aq < 4; ++aq) {
      const size_t ro = arow_off + (size_t)(aq * 16) * DIMS;
      if constexpr (BF16SRC) {
        const unsigned short* xr = xb + ro;
        a[aq].q[0] = *(const uint4*)(xr + k0);
        a[aq].q[1] = *(const uint4*)(xr + k0 + 16);
        __builtin_prefetch(xr + k0 + 128, 0, 0);   // global_prefetch_b8
      } else {
        const float* xr = xf + ro;
        float4 f0 = *(const float4*)(xr + k0);
        float4 f1 = *(const float4*)(xr + k0 + 4);
        float4 f2 = *(const float4*)(xr + k0 + 16);
        float4 f3 = *(const float4*)(xr + k0 + 20);
        a[aq].h[0] = f32_to_bf16(f0.x); a[aq].h[1] = f32_to_bf16(f0.y);
        a[aq].h[2] = f32_to_bf16(f0.z); a[aq].h[3] = f32_to_bf16(f0.w);
        a[aq].h[4] = f32_to_bf16(f1.x); a[aq].h[5] = f32_to_bf16(f1.y);
        a[aq].h[6] = f32_to_bf16(f1.z); a[aq].h[7] = f32_to_bf16(f1.w);
        a[aq].h[8]  = f32_to_bf16(f2.x); a[aq].h[9]  = f32_to_bf16(f2.y);
        a[aq].h[10] = f32_to_bf16(f2.z); a[aq].h[11] = f32_to_bf16(f2.w);
        a[aq].h[12] = f32_to_bf16(f3.x); a[aq].h[13] = f32_to_bf16(f3.y);
        a[aq].h[14] = f32_to_bf16(f3.z); a[aq].h[15] = f32_to_bf16(f3.w);
        __builtin_prefetch(xr + k0 + 128, 0, 0);
      }
    }

    // Prefetch next chunk into the other buffer, then drain this chunk's copies.
    if (kc + 1 < NKCH) {
      issue_btile(cur ^ 1, kc + 1);
      wait_asynccnt<4>();      // the 4 older copies (current buffer) have landed
    } else {
      wait_asynccnt<0>();
    }
    __syncthreads();           // current buffer visible block-wide

    // 4 B frags from LDS, each reused by 4 WMMAs (one dscnt wait per 4 WMMAs).
    const unsigned short* bt = &btile[cur][half * 8];
#pragma unroll
    for (int nt = 0; nt < 4; ++nt) {
      const unsigned short* br = bt + (size_t)(cg * 64 + nt * 16 + l15) * BROW;
      FragBF b;
      b.q[0] = *(const uint4*)(br);
      b.q[1] = *(const uint4*)(br + 16);
#pragma unroll
      for (int aq = 0; aq < 4; ++aq) {
        acc[aq * 4 + nt] = __builtin_amdgcn_wmma_f32_16x16x32_bf16(
            /*neg_a=*/false, a[aq].v, /*neg_b=*/false, b.v,
            /*c_mod=*/(short)0, acc[aq * 4 + nt], /*reuse_a=*/false, /*reuse_b=*/false);
      }
    }
    __syncthreads();           // all waves done reading before buffer is rewritten
  }

  // C layout: lane l holds column l&15; rows (l>=16?8:0) + vgpr(0..7).
  float bestv[4][8];
  int   besti[4][8];
#pragma unroll
  for (int aq = 0; aq < 4; ++aq)
#pragma unroll
    for (int i = 0; i < 8; ++i) { bestv[aq][i] = __builtin_inff(); besti[aq][i] = 0; }

#pragma unroll
  for (int nt = 0; nt < 4; ++nt) {
    int c = cg * 64 + nt * 16 + l15;
    float nrm = norms[c];
#pragma unroll
    for (int aq = 0; aq < 4; ++aq) {
#pragma unroll
      for (int i = 0; i < 8; ++i) {
        float s = nrm - 2.0f * acc[aq * 4 + nt][i];
        if (s < bestv[aq][i]) { bestv[aq][i] = s; besti[aq][i] = c; }  // increasing c
      }
    }
  }
  // cross-lane argmin within each 16-lane half (columns 0..15)
#pragma unroll
  for (int m = 1; m <= 8; m <<= 1) {
#pragma unroll
    for (int aq = 0; aq < 4; ++aq) {
#pragma unroll
      for (int i = 0; i < 8; ++i) {
        float ov = __shfl_xor(bestv[aq][i], m, WAVE);
        int   oi = __shfl_xor(besti[aq][i], m, WAVE);
        if (ov < bestv[aq][i] || (ov == bestv[aq][i] && oi < besti[aq][i])) {
          bestv[aq][i] = ov; besti[aq][i] = oi;
        }
      }
    }
  }
  // stash per-classgroup winners, then combine the 4 class groups per row
  if (l15 == 0) {
#pragma unroll
    for (int aq = 0; aq < 4; ++aq) {
#pragma unroll
      for (int i = 0; i < 8; ++i) {
        int rl = rg * 64 + aq * 16 + half * 8 + i;   // row within block
        sv[cg][rl] = bestv[aq][i];
        si[cg][rl] = besti[aq][i];
      }
    }
  }
  __syncthreads();
  if (t < RPB) {
    float bv = sv[0][t];
    int   bi = si[0][t];
#pragma unroll
    for (int g = 1; g < 4; ++g) {
      float ov = sv[g][t];
      int   oi = si[g][t];
      if (ov < bv || (ov == bv && oi < bi)) { bv = ov; bi = oi; }
    }
    labels[brow0 + t] = (unsigned int)bi;
  }
}

// ---------------- labels -> int64 output ------------------------------------
__global__ void labels_to_i64(const unsigned int* __restrict__ labels, long long* __restrict__ out) {
  int i = blockIdx.x * blockDim.x + threadIdx.x;
  out[i] = (long long)labels[i];
}

extern "C" void kernel_launch(void* const* d_in, const int* in_sizes, int n_in,
                              void* d_out, int out_size, void* d_ws, size_t ws_size,
                              hipStream_t stream) {
  (void)in_sizes; (void)n_in; (void)out_size;
  const float* features = (const float*)d_in[0];
  const float* logits   = (const float*)d_in[1];

  // Workspace layout (small region first; bf16 feature copy last, optional):
  char* ws = (char*)d_ws;
  size_t off = 0;
  float* sums = (float*)(ws + off);                off += (size_t)NCLASS * DIMS * 4;  // 1 MB
  float* counts = (float*)(ws + off);              off += (size_t)NCLASS * 4;         // 1 KB
  float* centroids = (float*)(ws + off);           off += (size_t)NCLASS * DIMS * 4;  // 1 MB
  float* norms = (float*)(ws + off);               off += (size_t)NCLASS * 4;         // 1 KB
  unsigned short* cbuf = (unsigned short*)(ws + off); off += (size_t)NCLASS * DIMS * 2; // 0.5 MB
  unsigned int* labels = (unsigned int*)(ws + off);   off += (size_t)NROWS * 4;       // 256 KB
  unsigned short* xb = (unsigned short*)(ws + off);
  size_t need_xb = off + (size_t)NROWS * DIMS * 2;                                    // +128 MB
  const bool use_bf16_copy = (ws_size >= need_xb);

  const int assign_blocks = NROWS / RPB;            // 512 blocks of 8 waves
  const int zero_iter_n   = NCLASS * DIMS + NCLASS; // sums + counts (contiguous)
  const int zero_init_n   = 2 * NCLASS * DIMS + NCLASS; // sums + counts + centroids

  if (use_bf16_copy) {
    convert_bf16<<<(NROWS * (DIMS / 8)) / 256, 256, 0, stream>>>(features, xb);
  }
  argmax_rows<<<NROWS / 8, 256, 0, stream>>>(logits, labels);
  zero_f32<<<(zero_init_n + 255) / 256, 256, 0, stream>>>(sums, zero_init_n);
  if (use_bf16_copy)
    accumulate_bf16<<<NROWS, 256, 0, stream>>>(xb, labels, sums, counts);
  else
    accumulate_f32<<<NROWS, 256, 0, stream>>>(features, labels, sums, counts);
  finalize_centroids<<<NCLASS, 256, 0, stream>>>(sums, counts, centroids, cbuf, norms);

  for (int it = 0; it < NITERS; ++it) {
    if (use_bf16_copy)
      assign_kernel<true><<<assign_blocks, 256, 0, stream>>>(xb, cbuf, norms, labels);
    else
      assign_kernel<false><<<assign_blocks, 256, 0, stream>>>(features, cbuf, norms, labels);
    zero_f32<<<(zero_iter_n + 255) / 256, 256, 0, stream>>>(sums, zero_iter_n);
    if (use_bf16_copy)
      accumulate_bf16<<<NROWS, 256, 0, stream>>>(xb, labels, sums, counts);
    else
      accumulate_f32<<<NROWS, 256, 0, stream>>>(features, labels, sums, counts);
    finalize_centroids<<<NCLASS, 256, 0, stream>>>(sums, counts, centroids, cbuf, norms);
  }

  if (use_bf16_copy)
    assign_kernel<true><<<assign_blocks, 256, 0, stream>>>(xb, cbuf, norms, labels);
  else
    assign_kernel<false><<<assign_blocks, 256, 0, stream>>>(features, cbuf, norms, labels);

  labels_to_i64<<<NROWS / 256, 256, 0, stream>>>(labels, (long long*)d_out);
}
